// Encoder_25366076850852
// MI455X (gfx1250) — matrix operand order repaired and checked by reading, not compiled, not run on hardware
//
#include <hip/hip_runtime.h>
#include <hip/hip_bf16.h>

// ---------------------------------------------------------------------------
// Types
// ---------------------------------------------------------------------------
typedef __attribute__((ext_vector_type(16))) __bf16 bf16x16;
typedef __attribute__((ext_vector_type(8)))  __bf16 bf16x8;
typedef __attribute__((ext_vector_type(8)))  float  f32x8;
typedef __attribute__((ext_vector_type(4)))  float  f32x4;
typedef __attribute__((ext_vector_type(2)))  unsigned u32x2;

// pack two fp32 -> two bf16 in one dword (lo = a, hi = b)
#if __has_builtin(__builtin_amdgcn_cvt_pk_bf16_f32)
static __device__ __forceinline__ unsigned pk2bf(float a, float b) {
  auto v = __builtin_amdgcn_cvt_pk_bf16_f32(a, b);   // single v_cvt_pk_bf16_f32
  return __builtin_bit_cast(unsigned, v);
}
#else
static __device__ __forceinline__ unsigned pk2bf(float a, float b) {
  unsigned ua = __builtin_bit_cast(unsigned, a);
  unsigned ub = __builtin_bit_cast(unsigned, b);
  ua += 0x7FFFu + ((ua >> 16) & 1u);                 // round-to-nearest-even
  ub += 0x7FFFu + ((ub >> 16) & 1u);
  return (ua >> 16) | (ub & 0xFFFF0000u);
}
#endif

// ---------------------------------------------------------------------------
// bf16 WMMA GEMM: C[M,N] = A[M,K] @ B[K,N], fp32 in/out, bf16 multiply,
// fp32 accumulate.  Block tile 128x64, 8 waves (16 rows each), K step 64
// (2 x 16x16x32 WMMA sub-steps -> 8 WMMA per barrier pair).
// Requires M%128==0, N%64==0, K%64==0 (true for every GEMM in this pipeline).
// ---------------------------------------------------------------------------
#define BM 128
#define BN 64
#define BK 64
#define LDSA 72   // bf16 elems per A row in LDS (64 + 8 pad, 144B, 16B aligned)
#define LDSB 72   // bf16 elems per B column in LDS

__global__ __launch_bounds__(256) void gemm_bf16_wmma(
    const float* __restrict__ A, const float* __restrict__ B,
    float* __restrict__ C, int M, int N, int K) {
  __shared__ __bf16 As[BM * LDSA];
  __shared__ __bf16 Bs[BN * LDSB];

  const int tid  = threadIdx.x;
  const int wave = tid >> 5;        // 0..7
  const int lane = tid & 31;
  const int tileM = blockIdx.y * BM;
  const int tileN = blockIdx.x * BN;

  // ---- hoist per-thread row pointers out of the K loop ----
  const float* apt[8];              // A tile: 128 rows x 64 cols
#pragma unroll
  for (int i = 0; i < 8; ++i) {
    int l = tid + i * 256;                   // 0..2047
    int r = l >> 4;                          // 0..127
    int c4 = (l & 15) << 2;                  // 0..60
    apt[i] = A + (size_t)(tileM + r) * K + c4;
  }
  const float* bpt[2];              // B tile: 64 K-rows x 64 cols, K-paired
#pragma unroll
  for (int i = 0; i < 2; ++i) {
    int l = tid + i * 256;                   // 0..511
    int kp = l >> 4;                         // K pair 0..31
    int c4 = (l & 15) << 2;                  // col 0..60
    bpt[i] = B + (size_t)(2 * kp) * N + tileN + c4;
  }
  const float* apre = A + (size_t)(tileM + (tid >> 1)) * K + ((tid & 1) << 5);
  const float* bpre = B + (size_t)(tid >> 1) * N + tileN + ((tid & 1) << 5);

  f32x8 acc[4] = {};                // wave: rows [wave*16,+16) x cols [0,64)

  size_t boff = 0;                  // = kk * N
  for (int kk = 0; kk < K; kk += BK, boff += (size_t)BK * N) {
    // ---- stage 1: issue ALL global tile loads (clause, overlapping) ----
    f32x4 av[8];
#pragma unroll
    for (int i = 0; i < 8; ++i) av[i] = *(const f32x4*)(apt[i] + kk);
    f32x4 bv0[2], bv1[2];
#pragma unroll
    for (int i = 0; i < 2; ++i) {
      const float* p = bpt[i] + boff;
      bv0[i] = *(const f32x4*)p;             // K row 2*kp
      bv1[i] = *(const f32x4*)(p + N);       // K row 2*kp+1
    }
    // ---- prefetch next K tile: exactly one 128B line per thread ----
    if (kk + BK < K) {
      __builtin_prefetch(apre + kk + BK, 0, 1);
      if (tid < 128)
        __builtin_prefetch(bpre + boff + (size_t)BK * N, 0, 1);
    }
    // ---- stage 2: convert + store to LDS ----
#pragma unroll
    for (int i = 0; i < 8; ++i) {
      int l = tid + i * 256;
      int r = l >> 4;
      int c4 = (l & 15) << 2;
      u32x2 p;
      p.x = pk2bf(av[i].x, av[i].y);
      p.y = pk2bf(av[i].z, av[i].w);
      *(u32x2*)&As[r * LDSA + c4] = p;       // single ds_store_b64
    }
#pragma unroll
    for (int i = 0; i < 2; ++i) {            // B: pack K-pair per column
      int l = tid + i * 256;
      int kp = l >> 4;
      int c4 = (l & 15) << 2;
#pragma unroll
      for (int j = 0; j < 4; ++j) {
        unsigned pk = pk2bf(bv0[i][j], bv1[i][j]);
        *(unsigned*)&Bs[(c4 + j) * LDSB + 2 * kp] = pk;   // ds_store_b32
      }
    }
    __syncthreads();

    // ---- stage 3: 2 K-substeps x 4 N-tiles = 8 WMMA ----
    const int arow  = wave * 16 + (lane & 15);
    const int akoff = (lane < 16) ? 0 : 8;     // A: lane<16 K{0..7,16..23}
    const int bkoff = (lane < 16) ? 0 : 8;     // B: lane<16 K{0..7,16..23}
#pragma unroll
    for (int ks = 0; ks < BK; ks += 32) {
      bf16x16 afrag;
      *((bf16x8*)&afrag)     = *(const bf16x8*)&As[arow * LDSA + ks + akoff];
      *((bf16x8*)&afrag + 1) = *(const bf16x8*)&As[arow * LDSA + ks + akoff + 16];
#pragma unroll
      for (int j = 0; j < 4; ++j) {
        const int bcol = j * 16 + (lane & 15);
        bf16x16 bfrag;
        *((bf16x8*)&bfrag)     = *(const bf16x8*)&Bs[bcol * LDSB + ks + bkoff];
        *((bf16x8*)&bfrag + 1) = *(const bf16x8*)&Bs[bcol * LDSB + ks + bkoff + 16];
        acc[j] = __builtin_amdgcn_wmma_f32_16x16x32_bf16(
            false, afrag, false, bfrag, (short)0, acc[j], false, false);
      }
    }
    __syncthreads();
  }

  // ---- store D: VGPR e -> row (wave*16 + e + (lane<16?0:8)), col (lane&15) ----
  const int rbase = tileM + wave * 16 + ((lane < 16) ? 0 : 8);
  const int cb    = tileN + (lane & 15);
#pragma unroll
  for (int j = 0; j < 4; ++j)
#pragma unroll
    for (int e = 0; e < 8; ++e)
      C[(size_t)(rbase + e) * N + cb + j * 16] = acc[j][e];
}

// ---------------------------------------------------------------------------
// Graph / elementwise kernels
// ---------------------------------------------------------------------------
__global__ __launch_bounds__(256) void k_fill(float* p, float v, int n) {
  int i = blockIdx.x * 256 + threadIdx.x;
  if (i < n) p[i] = v;
}

__global__ __launch_bounds__(256) void k_deg_scatter(
    const int* __restrict__ dst, const float* __restrict__ w, float* deg, int E) {
  int e = blockIdx.x * 256 + threadIdx.x;
  if (e < E) atomicAdd(&deg[dst[e]], w[e]);
}

__global__ __launch_bounds__(256) void k_rsqrt(const float* deg, float* dis, int n) {
  int i = blockIdx.x * 256 + threadIdx.x;
  if (i < n) dis[i] = rsqrtf(deg[i]);          // deg >= 1 (self loop) so finite
}

__global__ __launch_bounds__(256) void k_norm(
    const int* __restrict__ src, const int* __restrict__ dst,
    const float* __restrict__ w, const float* __restrict__ dis,
    float* nrm, int E) {
  int e = blockIdx.x * 256 + threadIdx.x;
  if (e < E) nrm[e] = dis[src[e]] * w[e] * dis[dst[e]];
}

// out = bias[f] + h * dis[n]^2   (self-loop contribution + bias)
__global__ __launch_bounds__(256) void k_aggr_init(
    const float* __restrict__ h, const float* __restrict__ dis,
    const float* __restrict__ bias, float* out, int fmask, int fshift, long total) {
  long gid = (long)blockIdx.x * 256 + threadIdx.x;
  if (gid >= total) return;
  int n = (int)(gid >> fshift);
  int f = (int)gid & fmask;
  float dv = dis[n];
  out[gid] = bias[f] + h[gid] * dv * dv;
}

// one thread = (edge, 4 features): out[dst] += h[src] * norm[e]
__global__ __launch_bounds__(256) void k_edge_scatter(
    const float* __restrict__ h, const int* __restrict__ src,
    const int* __restrict__ dst, const float* __restrict__ nrm,
    float* out, int F, int fshift, long total) {
  long gid = (long)blockIdx.x * 256 + threadIdx.x;
  if (gid >= total) return;
  int e = (int)(gid >> fshift);
  int f = ((int)gid & ((1 << fshift) - 1)) << 2;
  int s = src[e], d = dst[e];
  float nw = nrm[e];
  f32x4 v = *(const f32x4*)(h + (size_t)s * F + f);
  float* o = out + (size_t)d * F + f;
  atomicAdd(o + 0, v.x * nw);
  atomicAdd(o + 1, v.y * nw);
  atomicAdd(o + 2, v.z * nw);
  atomicAdd(o + 3, v.w * nw);
}

__global__ __launch_bounds__(256) void k_prelu(
    float* x, const float* __restrict__ a, int fmask, long total) {
  long gid = (long)blockIdx.x * 256 + threadIdx.x;
  if (gid >= total) return;
  float v = x[gid];
  float al = a[(int)gid & fmask];
  x[gid] = (v >= 0.0f) ? v : al * v;
}

// Bcat[row, 0:64]=z, [64:128]=z_a, [128]=1.0 (rowsum probe), [129:192]=0
__global__ __launch_bounds__(192) void k_pack_bcat(
    const float* __restrict__ z, const float* __restrict__ za,
    float* bcat, int N) {
  int row = blockIdx.x;
  int col = threadIdx.x;           // 0..191
  float v;
  if (col < 64)       v = z[(size_t)row * 64 + col];
  else if (col < 128) v = za[(size_t)row * 64 + col - 64];
  else                v = (col == 128) ? 1.0f : 0.0f;
  bcat[(size_t)row * 192 + col] = v;
}

// from VScat[8192,192]: cols 0..63 -> g, 64..127 -> ga, col 128 = rowsum
__global__ __launch_bounds__(256) void k_readout(
    const float* __restrict__ vscat, float* g, float* ga, int N) {
  int wave = threadIdx.x >> 5, lane = threadIdx.x & 31;
  int row = blockIdx.x * 8 + wave;
  const float* v = vscat + (size_t)row * 192;
  float inv_rs = 1.0f / v[128];

  float x0 = v[lane * 2] * inv_rs;
  float x1 = v[lane * 2 + 1] * inv_rs;
  float ss = x0 * x0 + x1 * x1;
#pragma unroll
  for (int off = 16; off > 0; off >>= 1) ss += __shfl_xor(ss, off, 32);
  float inv = 1.0f / fmaxf(sqrtf(ss), 1e-12f);
  g[(size_t)row * 64 + lane * 2]     = 1.0f / (1.0f + __expf(-x0 * inv));
  g[(size_t)row * 64 + lane * 2 + 1] = 1.0f / (1.0f + __expf(-x1 * inv));

  float y0 = v[64 + lane * 2] * inv_rs;
  float y1 = v[64 + lane * 2 + 1] * inv_rs;
  float tt = y0 * y0 + y1 * y1;
#pragma unroll
  for (int off = 16; off > 0; off >>= 1) tt += __shfl_xor(tt, off, 32);
  float inva = 1.0f / fmaxf(sqrtf(tt), 1e-12f);
  ga[(size_t)row * 64 + lane * 2]     = 1.0f / (1.0f + __expf(-y0 * inva));
  ga[(size_t)row * 64 + lane * 2 + 1] = 1.0f / (1.0f + __expf(-y1 * inva));
}

// ret[:,0]=<t,g>, ret[:,1]=<ta,g>, ret_a[:,0]=<ta,ga>, ret_a[:,1]=<t,ga>  (+bb)
__global__ __launch_bounds__(256) void k_bilinear(
    const float* __restrict__ t, const float* __restrict__ ta,
    const float* __restrict__ g, const float* __restrict__ ga,
    const float* __restrict__ bb, float* ret, float* ret_a, int N) {
  int wave = threadIdx.x >> 5, lane = threadIdx.x & 31;
  int row = blockIdx.x * 8 + wave;
  size_t base = (size_t)row * 64 + lane * 2;
  float t0 = t[base],  t1 = t[base + 1];
  float u0 = ta[base], u1 = ta[base + 1];
  float g0 = g[base],  g1 = g[base + 1];
  float h0 = ga[base], h1 = ga[base + 1];
  float d0 = t0 * g0 + t1 * g1;
  float d1 = u0 * g0 + u1 * g1;
  float d2 = u0 * h0 + u1 * h1;
  float d3 = t0 * h0 + t1 * h1;
#pragma unroll
  for (int off = 16; off > 0; off >>= 1) {
    d0 += __shfl_xor(d0, off, 32);
    d1 += __shfl_xor(d1, off, 32);
    d2 += __shfl_xor(d2, off, 32);
    d3 += __shfl_xor(d3, off, 32);
  }
  if (lane == 0) {
    float b = bb[0];
    ret[row * 2]     = d0 + b;
    ret[row * 2 + 1] = d1 + b;
    ret_a[row * 2]     = d2 + b;
    ret_a[row * 2 + 1] = d3 + b;
  }
}

// ---------------------------------------------------------------------------
// Orchestration
// ---------------------------------------------------------------------------
extern "C" void kernel_launch(void* const* d_in, const int* in_sizes, int n_in,
                              void* d_out, int out_size, void* d_ws, size_t ws_size,
                              hipStream_t stream) {
  const float* expr   = (const float*)d_in[0];
  const float* expr_a = (const float*)d_in[1];
  const int*   elist  = (const int*)d_in[2];
  const float* ew     = (const float*)d_in[3];
  const float* mask   = (const float*)d_in[4];
  const float* W1     = (const float*)d_in[5];
  const float* b1     = (const float*)d_in[6];
  const float* a1     = (const float*)d_in[7];
  const float* W2     = (const float*)d_in[8];
  const float* b2     = (const float*)d_in[9];
  const float* a2     = (const float*)d_in[10];
  const float* Wb     = (const float*)d_in[11];  // [1,64,64] -> 64x64
  const float* bb     = (const float*)d_in[12];

  const int Nn = 8192, Fin = 1024, Hh = 512, Oo = 64;
  const int E = in_sizes[3];
  const int* src = elist;
  const int* dst = elist + E;

  float* ws = (float*)d_ws;
  float* deg   = ws;  ws += Nn;
  float* dis   = ws;  ws += Nn;
  float* nrm   = ws;  ws += E;
  float* bufH  = ws;  ws += (size_t)Nn * Hh;   // x@W1
  float* bufA  = ws;  ws += (size_t)Nn * Hh;   // layer-1 aggregate
  float* bufH2 = ws;  ws += (size_t)Nn * Oo;   // x2@W2
  float* za    = ws;  ws += (size_t)Nn * Oo;   // z for expr_a
  float* bcat  = ws;  ws += (size_t)Nn * 192;  // [z | za | ones | 0]
  float* vscat = ws;  ws += (size_t)Nn * 192;  // mask @ bcat
  float* g     = ws;  ws += (size_t)Nn * Oo;
  float* ga    = ws;  ws += (size_t)Nn * Oo;
  float* t     = ws;  ws += (size_t)Nn * Oo;
  float* ta    = ws;  ws += (size_t)Nn * Oo;

  float* z    = (float*)d_out;                 // [N,64]  (output 0)
  float* ret  = z + (size_t)Nn * Oo;           // [N,2]   (output 1)
  float* reta = ret + (size_t)Nn * 2;          // [N,2]   (output 2)

  // ---- degree / normalization (shared by all 4 propagations) ----
  k_fill<<<Nn / 256, 256, 0, stream>>>(deg, 1.0f, Nn);               // self loops
  k_deg_scatter<<<E / 256, 256, 0, stream>>>(dst, ew, deg, E);
  k_rsqrt<<<Nn / 256, 256, 0, stream>>>(deg, dis, Nn);
  k_norm<<<E / 256, 256, 0, stream>>>(src, dst, ew, dis, nrm, E);

  auto gemm = [&](const float* A, const float* B, float* C, int M, int Nc, int K) {
    dim3 grid(Nc / BN, M / BM);
    gemm_bf16_wmma<<<grid, 256, 0, stream>>>(A, B, C, M, Nc, K);
  };

  // ---- two-layer GCN encoder, run for expr then expr_a ----
  for (int pass = 0; pass < 2; ++pass) {
    const float* x = pass ? expr_a : expr;
    float* zo = pass ? za : z;

    gemm(x, W1, bufH, Nn, Hh, Fin);                                  // WMMA
    k_aggr_init<<<(Nn * Hh) / 256, 256, 0, stream>>>(bufH, dis, b1, bufA,
                                                     Hh - 1, 9, (long)Nn * Hh);
    k_edge_scatter<<<(unsigned)(((long)E * (Hh / 4)) / 256), 256, 0, stream>>>(
        bufH, src, dst, nrm, bufA, Hh, 7, (long)E * (Hh / 4));
    k_prelu<<<(Nn * Hh) / 256, 256, 0, stream>>>(bufA, a1, Hh - 1, (long)Nn * Hh);

    gemm(bufA, W2, bufH2, Nn, Oo, Hh);                               // WMMA
    k_aggr_init<<<(Nn * Oo) / 256, 256, 0, stream>>>(bufH2, dis, b2, zo,
                                                     Oo - 1, 6, (long)Nn * Oo);
    k_edge_scatter<<<(unsigned)(((long)E * (Oo / 4)) / 256), 256, 0, stream>>>(
        bufH2, src, dst, nrm, zo, Oo, 4, (long)E * (Oo / 4));
    k_prelu<<<(Nn * Oo) / 256, 256, 0, stream>>>(zo, a2, Oo - 1, (long)Nn * Oo);
  }

  // ---- readout: single pass over 256MB mask computes mask@z, mask@za AND
  //      rowsum(mask) (ones column) in one WMMA GEMM ----
  k_pack_bcat<<<Nn, 192, 0, stream>>>(z, za, bcat, Nn);
  gemm(mask, bcat, vscat, Nn, 192, Nn);                              // WMMA
  k_readout<<<Nn / 8, 256, 0, stream>>>(vscat, g, ga, Nn);

  // ---- bilinear discriminator ----
  gemm(z,  Wb, t,  Nn, Oo, Oo);                                      // WMMA
  gemm(za, Wb, ta, Nn, Oo, Oo);                                      // WMMA
  k_bilinear<<<Nn / 8, 256, 0, stream>>>(t, ta, g, ga, bb, ret, reta, Nn);
}
